// SelfAttention_33114197852163
// MI455X (gfx1250) — compile-verified, matrix operand order checked by reading, not run on hardware
//
#include <hip/hip_runtime.h>
#include <hip/hip_bf16.h>
#include <math.h>

typedef _Float16 half_t;
typedef __attribute__((ext_vector_type(16))) _Float16 v16h;
typedef __attribute__((ext_vector_type(8)))  float    v8f;

#define BATCH  2
#define SEQ    2048
#define DMODEL 2048
#define NHEAD  32
#define NKV    8
#define HDIM   64
#define DKV    (NKV*HDIM)      // 512
#define MROWS  (BATCH*SEQ)     // 4096
#define BK     32

// 16-bit WMMA operand fragment loader (wave32).
// ISA 16-bit A-matrix 16x32 layout: lane L holds row M=L%16; half h=2v+pos maps to
// K = (v<4 ? 2v : 16+2(v-4)) + pos + (L>=16 ? 8 : 0).  B (32x16) mirrors with lane=col N.
// Pattern is two contiguous 8-half chunks -> compiler emits 2x ds_load_b128 / global b128.
__device__ __forceinline__ v16h frag_ld(const half_t* src, int hi8) {
  v16h f;
#pragma unroll
  for (int v = 0; v < 8; ++v) {
    int k = ((v < 4) ? (2 * v) : (16 + 2 * (v - 4))) + hi8;
    f[2 * v]     = src[k];
    f[2 * v + 1] = src[k + 1];
  }
  return f;
}

__device__ __forceinline__ v8f wmma_f16(v16h a, v16h b, v8f c) {
  // args: (neg_a, A, neg_b, B, c_mod, C, reuse_a, reuse_b)
  return __builtin_amdgcn_wmma_f32_16x16x32_f16(false, a, false, b, (short)0, c,
                                                false, false);
}

// ---- CDNA5 async global->LDS copy (ASYNCcnt path, no VGPR bounce) -----------
__device__ __forceinline__ void async_ld_b128(unsigned lds_off, const void* gptr) {
  asm volatile("global_load_async_to_lds_b128 %0, %1, off"
               :: "v"(lds_off), "v"(gptr)
               : "memory");
}
template <int N>
__device__ __forceinline__ void wait_async() {
  asm volatile("s_wait_asynccnt %0" :: "i"(N) : "memory");
}
// Low 32 bits of a generic __shared__ pointer = LDS byte address (ISA 10.2).
__device__ __forceinline__ unsigned lds_off(const void* p) {
  return (unsigned)(size_t)p;
}

// ---------------------------------------------------------------------------
// fp32 -> f16 conversion
__global__ void cvt_f32_f16(const float* __restrict__ in, half_t* __restrict__ out,
                            long n) {
  long i = blockIdx.x * (long)blockDim.x + threadIdx.x;
  if (i < n) out[i] = (half_t)in[i];
}

// ---------------------------------------------------------------------------
// C[M,N] = A[M,K] * B[N,K]^T   (A,B f16; accum f32; out f16 or f32)
// Block tile 128x64, 8 waves, each wave: 16 rows x 64 cols (4 WMMA accs).
// Double-buffered LDS, tiles staged with global_load_async_to_lds_b128.
__global__ __launch_bounds__(256) void gemm_nt(const half_t* __restrict__ A,
                                               const half_t* __restrict__ Bw,
                                               half_t* __restrict__ Ch,
                                               float* __restrict__ Cf,
                                               int M, int N, int K, int outF32) {
  __shared__ __align__(16) half_t As[2][128][BK];
  __shared__ __align__(16) half_t Bs[2][64][BK];
  int t = threadIdx.x;
  int wave = t >> 5, lane = t & 31;
  int lm = lane & 15, hi8 = (lane & 16) ? 8 : 0;
  int m0 = blockIdx.y * 128, n0 = blockIdx.x * 64;

  v8f acc[4];
#pragma unroll
  for (int i = 0; i < 4; ++i) { v8f z = {}; acc[i] = z; }

  int rowA = t >> 1, kA = (t & 1) * 16;   // 256 thr * 2 b128 = 128x32 tile
  int rowB = t >> 2, kB = (t & 3) * 8;    // 256 thr * 1 b128 = 64x32 tile
  const half_t* gA = A + (size_t)(m0 + rowA) * K + kA;
  const half_t* gB = Bw + (size_t)(n0 + rowB) * K + kB;

  auto issue_tile = [&](int buf, int kk) {
    async_ld_b128(lds_off(&As[buf][rowA][kA]),     gA + kk);
    async_ld_b128(lds_off(&As[buf][rowA][kA + 8]), gA + kk + 8);
    async_ld_b128(lds_off(&Bs[buf][rowB][kB]),     gB + kk);
  };

  issue_tile(0, 0);
  int buf = 0;
  for (int kk = 0; kk < K; kk += BK) {
    if (kk + BK < K) {            // prefetch next tile, then wait for current
      issue_tile(buf ^ 1, kk + BK);
      wait_async<3>();            // 3 outstanding = only the prefetch pending
    } else {
      wait_async<0>();
    }
    __syncthreads();              // publish current tile to all waves

    v16h af = frag_ld(&As[buf][wave * 16 + lm][0], hi8);
#pragma unroll
    for (int nt = 0; nt < 4; ++nt) {
      v16h bf = frag_ld(&Bs[buf][nt * 16 + lm][0], hi8);
      acc[nt] = wmma_f16(af, bf, acc[nt]);
    }
    __syncthreads();              // all reads done before buffer is re-filled
    buf ^= 1;
  }

#pragma unroll
  for (int nt = 0; nt < 4; ++nt)
#pragma unroll
    for (int r = 0; r < 8; ++r) {
      size_t row = (size_t)(m0 + wave * 16 + r + hi8);
      size_t col = (size_t)(n0 + nt * 16 + lm);
      if (outF32) Cf[row * (size_t)N + col] = acc[nt][r];
      else        Ch[row * (size_t)N + col] = (half_t)acc[nt][r];
    }
}

// ---------------------------------------------------------------------------
// In-place RoPE over (rows, rowWidth) f16 buffer; pair p within head (32 pairs).
__global__ void rope_kernel(half_t* __restrict__ buf, int rowWidth, long npairs) {
  long idx = blockIdx.x * (long)blockDim.x + threadIdx.x;
  if (idx >= npairs) return;
  int halfw = rowWidth >> 1;
  long row = idx / halfw;
  int pc = (int)(idx - row * halfw);
  int p = pc & 31;                          // pair index within 64-dim head
  int pos = (int)(row & (SEQ - 1));
  // freq = theta^(-2p/64);  angle = pos * 2*pi * freq
  float freq = __expf(-(float)(2 * p) * (9.2103403720f / 64.0f));  // ln(1e4)
  float ang = (float)pos * 6.2831853072f * freq;
  float s, c;
  __sincosf(ang, &s, &c);
  half_t* e = buf + row * (long)rowWidth + 2 * pc;
  float a0 = (float)e[0], a1 = (float)e[1];
  e[0] = (half_t)(a0 * c - a1 * s);
  e[1] = (half_t)(a0 * s + a1 * c);
}

// ---------------------------------------------------------------------------
// V (b*S, DKV) -> Vt (b, h_kv, d, s) so P*V B-fragments load contiguously per lane.
__global__ void transpose_v(const half_t* __restrict__ Vb, half_t* __restrict__ Vt,
                            long n) {
  long i = blockIdx.x * (long)blockDim.x + threadIdx.x;
  if (i >= n) return;
  int c = (int)(i % DKV);
  long rs = i / DKV;                       // b*SEQ + s
  int s = (int)(rs & (SEQ - 1));
  int b = (int)(rs >> 11);                 // SEQ = 2^11
  int h = c >> 6, d = c & 63;
  Vt[(((long)(b * NKV + h) * HDIM + d) << 11) + s] = Vb[i];
}

// ---------------------------------------------------------------------------
// Flash attention: grid (S/128, NHEAD, BATCH), 8 waves/block, wave = 16 query rows.
__global__ __launch_bounds__(256) void flash_attn(const half_t* __restrict__ Qb,
                                                  const half_t* __restrict__ Kb,
                                                  const half_t* __restrict__ Vt,
                                                  half_t* __restrict__ Zb) {
  __shared__ __align__(16) half_t Pt[8][16][32];  // per-wave P transpose buffer
  int w = threadIdx.x >> 5, lane = threadIdx.x & 31;
  int lm = lane & 15, hi8 = (lane & 16) ? 8 : 0;
  int b = blockIdx.z, h = blockIdx.y;
  int hkv = h >> 2;                               // GQA group = 4
  int q0 = blockIdx.x * 128 + w * 16;

  // Q fragments (hd 0..31, 32..63) straight from global
  const half_t* qrow = Qb + (size_t)(b * SEQ + q0 + lm) * DMODEL + h * HDIM;
  v16h qf0 = frag_ld(qrow, hi8);
  v16h qf1 = frag_ld(qrow + 32, hi8);

  v8f O[4];
#pragma unroll
  for (int i = 0; i < 4; ++i) { v8f z = {}; O[i] = z; }
  float mrow[8], lrow[8];
#pragma unroll
  for (int r = 0; r < 8; ++r) { mrow[r] = -INFINITY; lrow[r] = 0.0f; }

  const float scale = 0.125f;  // 1/sqrt(64)
  const half_t* Kbase = Kb + (size_t)(b * SEQ) * DKV + hkv * HDIM;
  const half_t* Vbase = Vt + (size_t)(b * NKV + hkv) * HDIM * SEQ;
  int kcMax = (q0 + 15) >> 5;

  for (int kc = 0; kc <= kcMax; ++kc) {
    int k0 = kc << 5;
    v8f s0 = {}, s1 = {};
    {
      v16h kf;
      kf = frag_ld(Kbase + (size_t)(k0 + lm) * DKV, hi8);           s0 = wmma_f16(qf0, kf, s0);
      kf = frag_ld(Kbase + (size_t)(k0 + lm) * DKV + 32, hi8);      s0 = wmma_f16(qf1, kf, s0);
      kf = frag_ld(Kbase + (size_t)(k0 + 16 + lm) * DKV, hi8);      s1 = wmma_f16(qf0, kf, s1);
      kf = frag_ld(Kbase + (size_t)(k0 + 16 + lm) * DKV + 32, hi8); s1 = wmma_f16(qf1, kf, s1);
    }
    bool diag = (k0 + 31 > q0);
#pragma unroll
    for (int r = 0; r < 8; ++r) {
      float a = s0[r] * scale, bb = s1[r] * scale;
      if (diag) {
        int q = q0 + r + hi8;
        if (k0 + lm > q)      a  = -1e30f;
        if (k0 + 16 + lm > q) bb = -1e30f;
      }
      // row max across the 16 lanes holding this row
      float mx = fmaxf(a, bb);
      mx = fmaxf(mx, __shfl_xor(mx, 1, 16));
      mx = fmaxf(mx, __shfl_xor(mx, 2, 16));
      mx = fmaxf(mx, __shfl_xor(mx, 4, 16));
      mx = fmaxf(mx, __shfl_xor(mx, 8, 16));
      float mnew  = fmaxf(mrow[r], mx);
      float alpha = __expf(mrow[r] - mnew);
      float p0 = __expf(a - mnew), p1 = __expf(bb - mnew);
      float rs = p0 + p1;
      rs += __shfl_xor(rs, 1, 16);
      rs += __shfl_xor(rs, 2, 16);
      rs += __shfl_xor(rs, 4, 16);
      rs += __shfl_xor(rs, 8, 16);
      lrow[r] = lrow[r] * alpha + rs;
      mrow[r] = mnew;
#pragma unroll
      for (int nt = 0; nt < 4; ++nt) O[nt][r] *= alpha;
      // D-layout (lane=col,VGPR=row) -> LDS row-major for A-fragment reload
      Pt[w][r + hi8][lm]      = (half_t)p0;
      Pt[w][r + hi8][lm + 16] = (half_t)p1;
    }
    v16h pf = frag_ld(&Pt[w][lm][0], hi8);
#pragma unroll
    for (int nt = 0; nt < 4; ++nt) {
      v16h vf = frag_ld(Vbase + (size_t)(nt * 16 + lm) * SEQ + k0, hi8);
      O[nt] = wmma_f16(pf, vf, O[nt]);
    }
  }

#pragma unroll
  for (int nt = 0; nt < 4; ++nt)
#pragma unroll
    for (int r = 0; r < 8; ++r) {
      size_t row = (size_t)(b * SEQ + q0 + r + hi8);
      Zb[row * DMODEL + h * HDIM + nt * 16 + lm] = (half_t)(O[nt][r] / lrow[r]);
    }
}

// ---------------------------------------------------------------------------
extern "C" void kernel_launch(void* const* d_in, const int* in_sizes, int n_in,
                              void* d_out, int out_size, void* d_ws, size_t ws_size,
                              hipStream_t stream) {
  (void)in_sizes; (void)n_in; (void)out_size; (void)ws_size;
  const float* x  = (const float*)d_in[0];
  const float* Wq = (const float*)d_in[1];
  const float* Wk = (const float*)d_in[2];
  const float* Wv = (const float*)d_in[3];
  const float* Wo = (const float*)d_in[4];
  float* out = (float*)d_out;

  const long nX  = (long)MROWS * DMODEL;   // 8.4M halves
  const long nWq = (long)DMODEL * DMODEL;  // 4.2M
  const long nWk = (long)DKV * DMODEL;     // 1.05M
  const long nKV = (long)MROWS * DKV;      // 2.1M

  half_t* p   = (half_t*)d_ws;
  half_t* xh  = p; p += nX;
  half_t* Wqh = p; p += nWq;
  half_t* Wkh = p; p += nWk;
  half_t* Wvh = p; p += nWk;
  half_t* Woh = p; p += nWq;
  half_t* Qb  = p; p += nX;
  half_t* Kb  = p; p += nKV;
  half_t* Vb  = p; p += nKV;
  half_t* Vt  = p; p += nKV;
  half_t* Zb  = p; p += nX;

  auto cdiv = [](long a, long b) { return (unsigned)((a + b - 1) / b); };
  dim3 blk(256);

  cvt_f32_f16<<<cdiv(nX, 256), blk, 0, stream>>>(x, xh, nX);
  cvt_f32_f16<<<cdiv(nWq, 256), blk, 0, stream>>>(Wq, Wqh, nWq);
  cvt_f32_f16<<<cdiv(nWk, 256), blk, 0, stream>>>(Wk, Wkh, nWk);
  cvt_f32_f16<<<cdiv(nWk, 256), blk, 0, stream>>>(Wv, Wvh, nWk);
  cvt_f32_f16<<<cdiv(nWq, 256), blk, 0, stream>>>(Wo, Woh, nWq);

  gemm_nt<<<dim3(DMODEL / 64, MROWS / 128), blk, 0, stream>>>(
      xh, Wqh, Qb, nullptr, MROWS, DMODEL, DMODEL, 0);
  gemm_nt<<<dim3(DKV / 64, MROWS / 128), blk, 0, stream>>>(
      xh, Wkh, Kb, nullptr, MROWS, DKV, DMODEL, 0);
  gemm_nt<<<dim3(DKV / 64, MROWS / 128), blk, 0, stream>>>(
      xh, Wvh, Vb, nullptr, MROWS, DKV, DMODEL, 0);

  rope_kernel<<<cdiv(nX / 2, 256), blk, 0, stream>>>(Qb, DMODEL, nX / 2);
  rope_kernel<<<cdiv(nKV / 2, 256), blk, 0, stream>>>(Kb, DKV, nKV / 2);
  transpose_v<<<cdiv(nKV, 256), blk, 0, stream>>>(Vb, Vt, nKV);

  flash_attn<<<dim3(SEQ / 128, NHEAD, BATCH), blk, 0, stream>>>(Qb, Kb, Vt, Zb);

  gemm_nt<<<dim3(DMODEL / 64, MROWS / 128), blk, 0, stream>>>(
      Zb, Woh, nullptr, out, MROWS, DMODEL, DMODEL, 1);
}
